// SCExtractorRLEntity_51539607552893
// MI455X (gfx1250) — compile-verified, matrix operand order checked by reading, not run on hardware
//
#include <hip/hip_runtime.h>
#include <hip/hip_bf16.h>

typedef __attribute__((ext_vector_type(16))) __bf16 v16bf;
typedef __attribute__((ext_vector_type(8)))  __bf16 v8bf;
typedef __attribute__((ext_vector_type(8)))  float  v8f;

#define NP1   257      // N_SENT + 1
#define DD    2048     // D == HID
#define G4    8192     // 4*HID
#define NSIDE 128
#define SDIM  512
#define MPAD  272      // 17*16, padded rows of mem for GEMM
#define NEGV  (-1.0e6f)

#define DEC_BLOCKS  64
#define DEC_THREADS 256
#define DEC_NW ((DEC_BLOCKS*DEC_THREADS)/32)   // 512 waves
#define DEC_NT (DEC_BLOCKS*DEC_THREADS)        // 16384 threads

// ---------------- small helpers ----------------
__device__ __forceinline__ unsigned short f2bf(float f){
  unsigned u = __float_as_uint(f);
  u += 0x7FFFu + ((u >> 16) & 1u);           // round-to-nearest-even
  return (unsigned short)(u >> 16);
}
__device__ __forceinline__ float sigm(float x){ return 1.0f / (1.0f + __expf(-x)); }

__device__ __forceinline__ float wred32(float v){
  #pragma unroll
  for (int m = 16; m > 0; m >>= 1) v += __shfl_xor(v, m, 32);
  return v;
}

// 16x32 bf16 tile loader per ISA 7.12.2 (16-bit A-matrix 16x32 layout).
// Works for A (row0 = M-tile base) and pre-transposed B ([N,K] row-major).
__device__ __forceinline__ v16bf load16x32(const __bf16* base, int ld, int row0, int k0, int lane){
  int r = lane & 15, half = lane >> 4;
  const __bf16* p = base + (size_t)(row0 + r) * (size_t)ld + (size_t)(k0 + half * 8);
  v8bf lo = *(const v8bf*)p;          // K = k0 + half*8 .. +7
  v8bf hi = *(const v8bf*)(p + 16);   // K = k0 + half*8 + 16 .. +23
  v16bf o;
  #pragma unroll
  for (int i = 0; i < 8; ++i){ o[i] = lo[i]; o[i + 8] = hi[i]; }
  return o;
}

// A-matrix = single vector in row 0, rows 1..15 zero (matvec-as-WMMA).
__device__ __forceinline__ v16bf loadvec32(const __bf16* v, int k0, int lane){
  v16bf o = {};
  if ((lane & 15) == 0){
    int half = lane >> 4;
    const __bf16* p = v + k0 + half * 8;
    v8bf lo = *(const v8bf*)p;
    v8bf hi = *(const v8bf*)(p + 16);
    #pragma unroll
    for (int i = 0; i < 8; ++i){ o[i] = lo[i]; o[i + 8] = hi[i]; }
  }
  return o;
}

// One 16-wide output tile of out = W(bf16 [Nout,K]) * vec(bf16 [K]); f32 out.
__device__ __forceinline__ void mv16(const __bf16* W, const __bf16* vec, float* out,
                                     int n0, int K, int lane){
  v8f acc = {};
  for (int k0 = 0; k0 < K; k0 += 32){
    v16bf av = loadvec32(vec, k0, lane);
    v16bf bv = load16x32(W, K, n0, k0, lane);
    acc = __builtin_amdgcn_wmma_f32_16x16x32_bf16(false, av, false, bv, (short)0, acc, false, false);
  }
  if (lane < 16) out[n0 + lane] = acc[0];   // D row0 lives in elem0 of lanes 0..15
}

// ---------------- conversion / packing kernels ----------------
__global__ void k_f2bf(const float* __restrict__ s, unsigned short* __restrict__ d, int n){
  int t = blockIdx.x * blockDim.x + threadIdx.x;
  if (t < n) d[t] = f2bf(s[t]);
}
// dst[C][R] = src[R][C] (bf16): gives [out,in]-major weights for vec@W matvecs / GEMM B.
__global__ void k_f2bf_T(const float* __restrict__ s, unsigned short* __restrict__ d, int R, int C){
  int t = blockIdx.x * blockDim.x + threadIdx.x;
  if (t >= R * C) return;
  int r = t / C, c = t - r * C;
  d[(size_t)c * R + r] = f2bf(s[t]);
}
// memA[272][2048] = concat(attn_mem, stop, zero-pad) in bf16
__global__ void k_pack_mem(const float* __restrict__ attn_mem, const float* __restrict__ stop,
                           unsigned short* __restrict__ memA){
  int t = blockIdx.x * blockDim.x + threadIdx.x;
  if (t >= MPAD * DD) return;
  int r = t >> 11, j = t & (DD - 1);
  float v = (r < 256) ? attn_mem[t] : (r == 256 ? stop[j] : 0.0f);
  memA[t] = f2bf(v);
}

__global__ void k_init(const float* ih, const float* ic, const float* ii,
                       float* h, float* c, float* x,
                       unsigned short* hb, unsigned short* xb,
                       int* mask, int* flags, unsigned* bcnt, unsigned* bgen){
  int t = blockIdx.x * blockDim.x + threadIdx.x;
  if (t < DD){ h[t] = ih[t]; c[t] = ic[t]; x[t] = ii[t]; hb[t] = f2bf(ih[t]); xb[t] = f2bf(ii[t]); }
  if (t < NP1) mask[t] = 0;
  if (t == 0){ flags[0] = 0; flags[1] = 0; flags[2] = 0; flags[3] = 0; *bcnt = 0u; *bgen = 0u; }
}

// ---------------- WMMA GEMM: C[M,N] = A[Mpad,K](bf16) * B[N,K](bf16, pre-transposed) ----------------
__global__ void __launch_bounds__(128) k_gemm_bf16(const unsigned short* __restrict__ Au,
                                                   const unsigned short* __restrict__ Bu,
                                                   float* __restrict__ C, int M, int N, int K){
  int wid  = blockIdx.x * 4 + (threadIdx.x >> 5);
  int lane = threadIdx.x & 31;
  int tilesN = N >> 4;
  int tilesM = (M + 15) >> 4;
  if (wid >= tilesM * tilesN) return;
  int mt = wid / tilesN, nt = wid - mt * tilesN;
  int m0 = mt << 4, n0 = nt << 4;
  const __bf16* A = (const __bf16*)Au;
  const __bf16* B = (const __bf16*)Bu;
  v8f acc = {};
  for (int k0 = 0; k0 < K; k0 += 32){
    v16bf a = load16x32(A, K, m0, k0, lane);
    v16bf b = load16x32(B, K, n0, k0, lane);
    acc = __builtin_amdgcn_wmma_f32_16x16x32_bf16(false, a, false, b, (short)0, acc, false, false);
  }
  int col   = n0 + (lane & 15);
  int rbase = m0 + (lane >> 4) * 8;
  #pragma unroll
  for (int e = 0; e < 8; ++e){
    int r = rbase + e;
    if (r < M) C[(size_t)r * N + col] = acc[e];
  }
}

// ---------------- persistent decode kernel ----------------
struct DecArgs {
  const unsigned short *Wih, *Whh, *hopWqT, *sideWqT, *attnWqT, *attnWsT;
  const float *b_ih, *b_hh, *attn_v, *hop_v, *side_v;
  const float *attn_mem, *stop;
  const float *attnF, *hopF, *sideF;
  float *h, *c, *x, *q, *qp, *qa, *sa, *sidee, *hnew, *cnew, *gates, *s_hop, *s_side, *masked;
  unsigned short *xb, *hb, *qb, *seb;
  int *mask, *flags;             // flags: [0]=done [1]=ext [2]=active [3]=old_done
  unsigned *bcnt, *bgen;
  float *scores_out; int *ext_out;
};

__device__ __forceinline__ void gbar(unsigned* cnt, unsigned* gen){
  __threadfence();                                    // release my writes to device scope
  __syncthreads();
  if (threadIdx.x == 0){
    unsigned g = __hip_atomic_load(gen, __ATOMIC_RELAXED, __HIP_MEMORY_SCOPE_AGENT);
    unsigned a = __hip_atomic_fetch_add(cnt, 1u, __ATOMIC_ACQ_REL, __HIP_MEMORY_SCOPE_AGENT);
    if (a == DEC_BLOCKS - 1){
      __hip_atomic_store(cnt, 0u, __ATOMIC_RELAXED, __HIP_MEMORY_SCOPE_AGENT);
      __hip_atomic_store(gen, g + 1u, __ATOMIC_RELEASE, __HIP_MEMORY_SCOPE_AGENT);
    } else {
      while (__hip_atomic_load(gen, __ATOMIC_ACQUIRE, __HIP_MEMORY_SCOPE_AGENT) == g)
        __builtin_amdgcn_s_sleep(2);
    }
  }
  __syncthreads();
  __threadfence();                                    // acquire: invalidate stale L0 lines
}

__global__ void __launch_bounds__(DEC_THREADS) k_decode(DecArgs P){
  const int tid  = threadIdx.x;
  const int gt   = blockIdx.x * DEC_THREADS + tid;
  const int gw   = gt >> 5;
  const int lane = tid & 31;

  const __bf16* Wih     = (const __bf16*)P.Wih;
  const __bf16* Whh     = (const __bf16*)P.Whh;
  const __bf16* hopWqT  = (const __bf16*)P.hopWqT;
  const __bf16* sideWqT = (const __bf16*)P.sideWqT;
  const __bf16* attnWqT = (const __bf16*)P.attnWqT;
  const __bf16* attnWsT = (const __bf16*)P.attnWsT;

  __shared__ float rb[256], rs[256];
  __shared__ int   rbi[256], rsi[256];

  for (int t = 0; t < 256; ++t){
    // ---- P1: gates = W_ih*x + W_hh*h + b  (WMMA matvec, 512 waves x 16 outs) ----
    {
      const __bf16* xb = (const __bf16*)P.xb;
      const __bf16* hb = (const __bf16*)P.hb;
      int n0 = gw << 4;                                // DEC_NW*16 == 8192
      v8f acc = {};
      for (int k0 = 0; k0 < DD; k0 += 32){
        v16bf a = loadvec32(xb, k0, lane);
        v16bf b = load16x32(Wih, DD, n0, k0, lane);
        acc = __builtin_amdgcn_wmma_f32_16x16x32_bf16(false, a, false, b, (short)0, acc, false, false);
      }
      for (int k0 = 0; k0 < DD; k0 += 32){
        v16bf a = loadvec32(hb, k0, lane);
        v16bf b = load16x32(Whh, DD, n0, k0, lane);
        acc = __builtin_amdgcn_wmma_f32_16x16x32_bf16(false, a, false, b, (short)0, acc, false, false);
      }
      if (lane < 16){
        int j = n0 + lane;
        P.gates[j] = acc[0] + P.b_ih[j] + P.b_hh[j];
      }
    }
    gbar(P.bcnt, P.bgen);

    // ---- P2: LSTM elementwise; q = h_new ----
    for (int j = gt; j < DD; j += DEC_NT){
      float gi = P.gates[j], gf = P.gates[DD + j], gg = P.gates[2*DD + j], go = P.gates[3*DD + j];
      float cn = sigm(gf) * P.c[j] + sigm(gi) * tanhf(gg);
      float hn = sigm(go) * tanhf(cn);
      P.hnew[j] = hn; P.cnew[j] = cn; P.q[j] = hn; P.qb[j] = f2bf(hn);
    }
    gbar(P.bcnt, P.bgen);

    // ---- P3: qp = hop_wq^T * q ----
    if (gw < DD/16) mv16(hopWqT, (const __bf16*)P.qb, P.qp, gw << 4, DD, lane);
    gbar(P.bcnt, P.bgen);

    // ---- P4: hop scores ----
    for (int n = gw; n < NP1; n += DEC_NW){
      const float* fr = P.hopF + (size_t)n * DD;
      float acc = 0.f;
      for (int j = lane; j < DD; j += 32) acc += tanhf(fr[j] + P.qp[j]) * P.hop_v[j];
      acc = wred32(acc);
      if (lane == 0) P.s_hop[n] = acc;
    }
    gbar(P.bcnt, P.bgen);

    // ---- P5: q = softmax(s_hop) @ hop_feat (softmax recomputed per thread) ----
    for (int j = gt; j < DD; j += DEC_NT){
      float mx = -INFINITY;
      for (int n = 0; n < NP1; ++n) mx = fmaxf(mx, P.s_hop[n]);
      float Z = 0.f, acc = 0.f;
      for (int n = 0; n < NP1; ++n){
        float e = __expf(P.s_hop[n] - mx);
        Z += e; acc += e * P.hopF[(size_t)n * DD + j];
      }
      float v = acc / Z;
      P.q[j] = v; P.qb[j] = f2bf(v);
    }
    gbar(P.bcnt, P.bgen);

    // ---- P6: qp = side_wq^T * q ----
    if (gw < DD/16) mv16(sideWqT, (const __bf16*)P.qb, P.qp, gw << 4, DD, lane);
    gbar(P.bcnt, P.bgen);

    // ---- P7: side scores ----
    for (int n = gw; n < NSIDE; n += DEC_NW){
      const float* fr = P.sideF + (size_t)n * DD;
      float acc = 0.f;
      for (int j = lane; j < DD; j += 32) acc += tanhf(fr[j] + P.qp[j]) * P.side_v[j];
      acc = wred32(acc);
      if (lane == 0) P.s_side[n] = acc;
    }
    gbar(P.bcnt, P.bgen);

    // ---- P8: side_e = softmax(s_side) @ side_feat ----
    for (int j = gt; j < DD; j += DEC_NT){
      float mx = -INFINITY;
      for (int n = 0; n < NSIDE; ++n) mx = fmaxf(mx, P.s_side[n]);
      float Z = 0.f, acc = 0.f;
      for (int n = 0; n < NSIDE; ++n){
        float e = __expf(P.s_side[n] - mx);
        Z += e; acc += e * P.sideF[(size_t)n * DD + j];
      }
      float v = acc / Z;
      P.sidee[j] = v; P.seb[j] = f2bf(v);
    }
    gbar(P.bcnt, P.bgen);

    // ---- P9: qa = attn_wq^T*q ; sa = attn_ws^T*side_e ----
    if (gw < 256){
      if (gw < 128) mv16(attnWqT, (const __bf16*)P.qb,  P.qa, gw << 4, DD, lane);
      else          mv16(attnWsT, (const __bf16*)P.seb, P.sa, (gw - 128) << 4, DD, lane);
    }
    gbar(P.bcnt, P.bgen);

    // ---- P10: masked scores ----
    for (int n = gw; n < NP1; n += DEC_NW){
      const float* fr = P.attnF + (size_t)n * DD;
      float acc = 0.f;
      for (int j = lane; j < DD; j += 32) acc += tanhf(fr[j] + P.qa[j] + P.sa[j]) * P.attn_v[j];
      acc = wred32(acc);
      if (lane == 0) P.masked[n] = P.mask[n] ? NEGV : acc;
    }
    gbar(P.bcnt, P.bgen);

    // ---- P11: top-2 + pointer logic (block 0) ----
    if (blockIdx.x == 0){
      float b = -INFINITY, s = -INFINITY; int bi = -1, si = -1;
      for (int n = tid; n < NP1; n += 256){
        float v = P.masked[n];
        if (v > b || (v == b && (unsigned)n < (unsigned)bi)){ s = b; si = bi; b = v; bi = n; }
        else if (v > s || (v == s && (unsigned)n < (unsigned)si)){ s = v; si = n; }
      }
      rb[tid] = b; rs[tid] = s; rbi[tid] = bi; rsi[tid] = si;
      __syncthreads();
      for (int off = 128; off > 0; off >>= 1){
        if (tid < off){
          float b1 = rb[tid],     s1 = rs[tid];     int i1 = rbi[tid],     j1 = rsi[tid];
          float b2 = rb[tid+off], s2 = rs[tid+off]; int i2 = rbi[tid+off], j2 = rsi[tid+off];
          float nb, ns; int nbi, nsi;
          bool f = (b1 > b2) || (b1 == b2 && (unsigned)i1 < (unsigned)i2);
          if (f){
            nb = b1; nbi = i1;
            bool g = (s1 > b2) || (s1 == b2 && (unsigned)j1 < (unsigned)i2);
            if (g){ ns = s1; nsi = j1; } else { ns = b2; nsi = i2; }
          } else {
            nb = b2; nbi = i2;
            bool g = (s2 > b1) || (s2 == b1 && (unsigned)j2 < (unsigned)i1);
            if (g){ ns = s2; nsi = j2; } else { ns = b1; nsi = i1; }
          }
          rb[tid] = nb; rs[tid] = ns; rbi[tid] = nbi; rsi[tid] = nsi;
        }
        __syncthreads();
      }
      if (tid == 0){
        int i1 = rbi[0], i2 = rsi[0];
        int od = P.flags[0];
        int e  = (t == 0 && i1 == 256) ? i2 : i1;
        int stop_now = (e == 256) && (t != 0);
        int act = (!od) && (!stop_now);
        if (act) P.mask[e] = 1;
        P.flags[3] = od; P.flags[1] = e; P.flags[2] = act; P.flags[0] = od | stop_now;
        P.ext_out[t] = act ? e : -1;
      }
      __syncthreads();
    }
    gbar(P.bcnt, P.bgen);

    // ---- P12: commit state + write score row ----
    {
      volatile int* fl = (volatile int*)P.flags;
      int act = fl[2], e = fl[1], od = fl[3];
      for (int j = gt; j < DD; j += DEC_NT){
        float hv = P.h[j], cv = P.c[j], xv = P.x[j];
        if (act){
          hv = P.hnew[j]; cv = P.cnew[j];
          xv = (e < 256) ? P.attn_mem[(size_t)e * DD + j] : P.stop[j];
        }
        P.h[j] = hv; P.c[j] = cv; P.x[j] = xv;
        P.hb[j] = f2bf(hv); P.xb[j] = f2bf(xv);
      }
      for (int n = gt; n < NP1; n += DEC_NT)
        P.scores_out[(size_t)t * NP1 + n] = od ? 0.0f : P.masked[n];
    }
    gbar(P.bcnt, P.bgen);
  }
}

// ---------------- host ----------------
extern "C" void kernel_launch(void* const* d_in, const int* in_sizes, int n_in,
                              void* d_out, int out_size, void* d_ws, size_t ws_size,
                              hipStream_t stream){
  const float* attn_mem = (const float*)d_in[0];
  const float* side_mem = (const float*)d_in[1];
  const float* stop     = (const float*)d_in[2];
  const float* init_h   = (const float*)d_in[3];
  const float* init_c   = (const float*)d_in[4];
  const float* init_i   = (const float*)d_in[5];
  const float* W_ih     = (const float*)d_in[6];
  const float* W_hh     = (const float*)d_in[7];
  const float* b_ih     = (const float*)d_in[8];
  const float* b_hh     = (const float*)d_in[9];
  const float* attn_wm  = (const float*)d_in[10];
  const float* attn_wq  = (const float*)d_in[11];
  const float* attn_ws  = (const float*)d_in[12];
  const float* attn_v   = (const float*)d_in[13];
  const float* hop_wm   = (const float*)d_in[14];
  const float* hop_wq   = (const float*)d_in[15];
  const float* hop_v    = (const float*)d_in[16];
  const float* side_wm  = (const float*)d_in[17];
  const float* side_wq  = (const float*)d_in[18];
  const float* side_v   = (const float*)d_in[19];

  size_t off = 0;
  auto take = [&](size_t bytes)->void*{
    void* p = (char*)d_ws + off;
    off += (bytes + 255) & ~(size_t)255;
    return p;
  };

  unsigned short* WihB    = (unsigned short*)take((size_t)G4 * DD * 2);
  unsigned short* WhhB    = (unsigned short*)take((size_t)G4 * DD * 2);
  unsigned short* hopWqT  = (unsigned short*)take((size_t)DD * DD * 2);
  unsigned short* sideWqT = (unsigned short*)take((size_t)DD * DD * 2);
  unsigned short* attnWqT = (unsigned short*)take((size_t)DD * DD * 2);
  unsigned short* attnWsT = (unsigned short*)take((size_t)DD * DD * 2);
  unsigned short* attnWmT = (unsigned short*)take((size_t)DD * DD * 2);
  unsigned short* hopWmT  = (unsigned short*)take((size_t)DD * DD * 2);
  unsigned short* sideWmT = (unsigned short*)take((size_t)DD * SDIM * 2);
  unsigned short* memA    = (unsigned short*)take((size_t)MPAD * DD * 2);
  unsigned short* sideMemA= (unsigned short*)take((size_t)NSIDE * SDIM * 2);
  float* attnF  = (float*)take((size_t)NP1 * DD * 4);
  float* hopF   = (float*)take((size_t)NP1 * DD * 4);
  float* sideF  = (float*)take((size_t)NSIDE * DD * 4);
  float* h      = (float*)take(DD * 4);
  float* c      = (float*)take(DD * 4);
  float* x      = (float*)take(DD * 4);
  float* q      = (float*)take(DD * 4);
  float* qp     = (float*)take(DD * 4);
  float* qa     = (float*)take(DD * 4);
  float* sa     = (float*)take(DD * 4);
  float* sidee  = (float*)take(DD * 4);
  float* hnew   = (float*)take(DD * 4);
  float* cnew   = (float*)take(DD * 4);
  float* gates  = (float*)take(G4 * 4);
  float* s_hop  = (float*)take(NP1 * 4);
  float* s_side = (float*)take(NSIDE * 4);
  float* masked = (float*)take(NP1 * 4);
  unsigned short* xb  = (unsigned short*)take(DD * 2);
  unsigned short* hb  = (unsigned short*)take(DD * 2);
  unsigned short* qb  = (unsigned short*)take(DD * 2);
  unsigned short* seb = (unsigned short*)take(DD * 2);
  int* mask  = (int*)take(NP1 * 4);
  int* flags = (int*)take(4 * 4);
  unsigned* bcnt = (unsigned*)take(4);
  unsigned* bgen = (unsigned*)take(4);
  (void)ws_size; (void)n_in; (void)in_sizes; (void)out_size;

  float* scores_out = (float*)d_out;
  int*   ext_out    = (int*)(scores_out + 256 * NP1);

  // init state + barrier
  k_init<<<8, 256, 0, stream>>>(init_h, init_c, init_i, h, c, x, hb, xb, mask, flags, bcnt, bgen);

  // bf16 weight conversions ([out,in]-major already for W_ih/W_hh)
  k_f2bf<<<(G4*DD + 255)/256, 256, 0, stream>>>(W_ih, WihB, G4*DD);
  k_f2bf<<<(G4*DD + 255)/256, 256, 0, stream>>>(W_hh, WhhB, G4*DD);
  // transposed bf16 for all vec@W matrices and GEMM B operands
  k_f2bf_T<<<(DD*DD + 255)/256, 256, 0, stream>>>(hop_wq,  hopWqT,  DD, DD);
  k_f2bf_T<<<(DD*DD + 255)/256, 256, 0, stream>>>(side_wq, sideWqT, DD, DD);
  k_f2bf_T<<<(DD*DD + 255)/256, 256, 0, stream>>>(attn_wq, attnWqT, DD, DD);
  k_f2bf_T<<<(DD*DD + 255)/256, 256, 0, stream>>>(attn_ws, attnWsT, DD, DD);
  k_f2bf_T<<<(DD*DD + 255)/256, 256, 0, stream>>>(attn_wm, attnWmT, DD, DD);
  k_f2bf_T<<<(DD*DD + 255)/256, 256, 0, stream>>>(hop_wm,  hopWmT,  DD, DD);
  k_f2bf_T<<<(SDIM*DD + 255)/256, 256, 0, stream>>>(side_wm, sideWmT, SDIM, DD);
  // activations to bf16
  k_pack_mem<<<(MPAD*DD + 255)/256, 256, 0, stream>>>(attn_mem, stop, memA);
  k_f2bf<<<(NSIDE*SDIM + 255)/256, 256, 0, stream>>>(side_mem, sideMemA, NSIDE*SDIM);

  // feature precompute GEMMs (WMMA): feat = mem @ wm ; side_feat = side_mem @ side_wm
  {
    int tiles = ((NP1 + 15)/16) * (DD/16);                 // 17*128
    k_gemm_bf16<<<(tiles + 3)/4, 128, 0, stream>>>(memA, attnWmT, attnF, NP1, DD, DD);
    k_gemm_bf16<<<(tiles + 3)/4, 128, 0, stream>>>(memA, hopWmT,  hopF,  NP1, DD, DD);
    int tilesS = (NSIDE/16) * (DD/16);                     // 8*128
    k_gemm_bf16<<<(tilesS + 3)/4, 128, 0, stream>>>(sideMemA, sideWmT, sideF, NSIDE, DD, SDIM);
  }

  // persistent sequential decode
  DecArgs P;
  P.Wih = WihB; P.Whh = WhhB; P.hopWqT = hopWqT; P.sideWqT = sideWqT;
  P.attnWqT = attnWqT; P.attnWsT = attnWsT;
  P.b_ih = b_ih; P.b_hh = b_hh; P.attn_v = attn_v; P.hop_v = hop_v; P.side_v = side_v;
  P.attn_mem = attn_mem; P.stop = stop;
  P.attnF = attnF; P.hopF = hopF; P.sideF = sideF;
  P.h = h; P.c = c; P.x = x; P.q = q; P.qp = qp; P.qa = qa; P.sa = sa; P.sidee = sidee;
  P.hnew = hnew; P.cnew = cnew; P.gates = gates;
  P.s_hop = s_hop; P.s_side = s_side; P.masked = masked;
  P.xb = xb; P.hb = hb; P.qb = qb; P.seb = seb;
  P.mask = mask; P.flags = flags; P.bcnt = bcnt; P.bgen = bgen;
  P.scores_out = scores_out; P.ext_out = ext_out;
  k_decode<<<DEC_BLOCKS, DEC_THREADS, 0, stream>>>(P);
}
